// KAN_63230508532179
// MI455X (gfx1250) — compile-verified, matrix operand order checked by reading, not run on hardware
//
#include <hip/hip_runtime.h>
#include <hip/hip_bf16.h>

typedef __attribute__((ext_vector_type(16))) _Float16 v16h;
typedef __attribute__((ext_vector_type(8)))  _Float16 v8h;
typedef __attribute__((ext_vector_type(8)))  float    v8f;
typedef int v4i __attribute__((vector_size(4 * sizeof(int))));  // b128 payload

#define B_SZ 4096
#define N_BR 1000
#define H1D  16
#define H2D  8
#define C1D  256
#define C2D  128
#define OUTD 300
#define KF   (N_BR * H2D)   // 8000

// CDNA5 async global->LDS path (guarded: falls back to load+ds_store if the
// toolchain lacks the builtins; __has_builtin keeps both toolchains compiling)
#if __has_builtin(__builtin_amdgcn_global_load_async_to_lds_b128) && \
    __has_builtin(__builtin_amdgcn_s_wait_asynccnt)
#define USE_ASYNC_LDS 1
#else
#define USE_ASYNC_LDS 0
#endif

#define AS1 __attribute__((address_space(1)))
#define AS3 __attribute__((address_space(3)))

__device__ __forceinline__ float gelu_f(float v) {
    // exact GeLU: 0.5*x*(1+erf(x/sqrt(2)))
    return 0.5f * v * (1.0f + erff(v * 0.7071067811865475f));
}

// branch-free tanh: (e^2x - 1) * rcp(e^2x + 1), clamped so e stays finite
__device__ __forceinline__ float tanh_fast(float v) {
    v = fminf(fmaxf(v, -10.0f), 10.0f);
    const float e = __expf(2.0f * v);
    return (e - 1.0f) * __builtin_amdgcn_rcpf(e + 1.0f);
}

__device__ __forceinline__ v16h zero16() {
    v16h r;
#pragma unroll
    for (int i = 0; i < 16; ++i) r[i] = (_Float16)0.0f;
    return r;
}

// Build a 16-bit A/B WMMA fragment for this lane from K-contiguous memory.
// Lane group g (lane>>4): elements 0..7 = K[8g..8g+7], elements 8..15 =
// K[16+8g..16+8g+7].  `p` must already point at row_base + 8*g.
__device__ __forceinline__ v16h load_frag(const _Float16* p) {
    v8h lo = *(const v8h*)(p);
    v8h hi = *(const v8h*)(p + 16);
    v16h r;
#pragma unroll
    for (int i = 0; i < 8; ++i) { r[i] = lo[i]; r[8 + i] = hi[i]; }
    return r;
}

// ---------------------------------------------------------------------------
// fp32 -> fp16 weight conversion
// ---------------------------------------------------------------------------
__global__ void cvt_f16_kernel(const float* __restrict__ src,
                               _Float16* __restrict__ dst, int n) {
    int i = blockIdx.x * blockDim.x + threadIdx.x;
    if (i < n) dst[i] = (_Float16)src[i];
}

// ---------------------------------------------------------------------------
// Branch stage: h = gelu(x*bw1+bb1) [16], f = tanh(h @ bw2^T + bb2) [8]
// One wave = 16 batch rows x 1 branch via one zero-padded WMMA
// (M=16 batch, K=16 hidden padded to 32, N=8 padded to 16).
// Results staged in LDS, written back as coalesced b128 rows.
// ---------------------------------------------------------------------------
__global__ void branch_kernel(const float* __restrict__ x,     // [B, N_BR]
                              const float* __restrict__ bw1,   // [N_BR, 16]
                              const float* __restrict__ bb1,   // [N_BR, 16]
                              const float* __restrict__ bw2,   // [N_BR, 8, 16]
                              const float* __restrict__ bb2,   // [N_BR, 8]
                              _Float16* __restrict__ f) {      // [B, 8000] f16
    __shared__ _Float16 fs[16 * 64];          // 16 rows x (8 branches * 8 out)

    const int lane = threadIdx.x & 31;
    const int wid  = threadIdx.x >> 5;
    const int n    = blockIdx.y * 8 + wid;    // branch index (125*8 = 1000)
    const int b0   = blockIdx.x * 16;         // batch row base
    const int m    = lane & 15;
    const int g    = lane >> 4;

    // ---- A fragment: rows = batch, K = hidden (16, padded to 32) ----
    const float xv = x[(size_t)(b0 + m) * N_BR + n];
    v16h a = zero16();
#pragma unroll
    for (int i = 0; i < 8; ++i) {
        const int k = 8 * g + i;                           // K in [0,16)
        const float h = gelu_f(xv * bw1[n * H1D + k] + bb1[n * H1D + k]);
        a[i] = (_Float16)h;                                // K 16..31 stay zero
    }

    // ---- B fragment: B[k, col] = bw2[n, col, k]; cols 8..15 / K>=16 zero ----
    const int col = m;
    const int cc  = (col < H2D) ? col : 0;                 // clamp (in-bounds)
    v16h b = zero16();
#pragma unroll
    for (int i = 0; i < 8; ++i) {
        const int k = 8 * g + i;
        const float w = bw2[((size_t)n * H2D + cc) * H1D + k];
        b[i] = (col < H2D) ? (_Float16)w : (_Float16)0.0f;
    }

    v8f c = {};
    c = __builtin_amdgcn_wmma_f32_16x16x32_f16(false, a, false, b,
                                               (short)0, c, false, false);

    if (col < H2D) {
        const float bias = bb2[n * H2D + col];
#pragma unroll
        for (int r = 0; r < 8; ++r) {
            const int row = r + 8 * g;        // lanes 16-31 cover M 8..15
            fs[row * 64 + wid * 8 + col] = (_Float16)tanh_fast(c[r] + bias);
        }
    }
    __syncthreads();

    // coalesced write: 16 rows x 128B, one b128 chunk per thread (t < 128)
    const int t = threadIdx.x;
    if (t < 128) {
        const int row = t >> 3, ch = t & 7;
        *(v8h*)(f + (size_t)(b0 + row) * KF + (size_t)blockIdx.y * 64 + ch * 8) =
            *(const v8h*)&fs[row * 64 + ch * 8];
    }
}

// ---------------------------------------------------------------------------
// WMMA GEMM: out = act(A[M,K] @ W[N,K]^T + bias)
// Block = 256 threads = 8 waves; tile 128(M) x 64(N); wave tile 32x32.
// B K-chunk staged in LDS (async global->LDS when available), fragments read
// via ds_load_b128; epilogue staged in LDS for coalesced b128 stores.
// ---------------------------------------------------------------------------
template <bool GELU_ACT, bool OUT_F32>
__global__ void gemm_wmma_kernel(const _Float16* __restrict__ A,   // [M][K]
                                 const _Float16* __restrict__ W,   // [Nvalid][K]
                                 const float* __restrict__ bias,   // [Nvalid]
                                 void* __restrict__ out,           // [M][Nout]
                                 int M, int K, int Nvalid, int Nout) {
    __shared__ _Float16 Bs[64 * 32];          // B K-chunk stage (4 KB)
    __shared__ char     smem[128 * 64 * 4];   // epilogue stage (32 KB)

    const int t    = threadIdx.x;
    const int lane = t & 31;
    const int wid  = t >> 5;
    const int wm   = wid & 3;                 // 4 M-subtiles of 32 rows
    const int wn   = wid >> 2;                // 2 N-subtiles of 32 cols
    const int m_base = blockIdx.x * 128 + wm * 32;
    const int n_blk  = blockIdx.y * 64;
    const int l15 = lane & 15;
    const int g   = lane >> 4;

    const int ncol0 = n_blk + wn * 32 + l15;
    const int ncol1 = n_blk + wn * 32 + 16 + l15;
    const bool b0ok = (ncol0 < Nvalid);
    const bool b1ok = (ncol1 < Nvalid);

    // A fragment pointers (direct global; rows always in range)
    const _Float16* a0p = A + (size_t)(m_base + l15) * K + 8 * g;
    const _Float16* a1p = A + (size_t)(m_base + 16 + l15) * K + 8 * g;

    // cooperative B stage mapping: thread -> (col 0..63, 16B sub-chunk 0..3)
    const int scol  = t >> 2;
    const int ssub  = t & 3;
    const int gcol0 = n_blk + scol;
    const int gcol  = (gcol0 < Nvalid) ? gcol0 : (Nvalid - 1);  // clamp; unused
    const _Float16* gsrc = W + (size_t)gcol * K + ssub * 8;     //   cols dropped
    _Float16* ldst = &Bs[scol * 32 + ssub * 8];

    // LDS fragment read pointers
    const _Float16* bs0 = &Bs[(wn * 32 + l15) * 32 + 8 * g];
    const _Float16* bs1 = &Bs[(wn * 32 + 16 + l15) * 32 + 8 * g];

    v8f c00 = {}, c01 = {}, c10 = {}, c11 = {};

    for (int k0 = 0; k0 < K; k0 += 32) {
#if USE_ASYNC_LDS
        __builtin_amdgcn_global_load_async_to_lds_b128(
            (AS1 v4i*)(unsigned long long)(uintptr_t)(gsrc + k0),
            (AS3 v4i*)(unsigned int)(uintptr_t)ldst, 0, 0);
        __builtin_amdgcn_s_wait_asynccnt(0);
#else
        *(v8h*)ldst = *(const v8h*)(gsrc + k0);
#endif
        __syncthreads();

        if (k0 + 32 < K) {                    // global_prefetch_b8 next A chunk
            __builtin_prefetch(a0p + k0 + 32, 0, 1);
            __builtin_prefetch(a1p + k0 + 32, 0, 1);
        }
        const v16h a0 = load_frag(a0p + k0);
        const v16h a1 = load_frag(a1p + k0);
        const v16h b0 = load_frag(bs0);       // ds_load_b128 x2
        const v16h b1 = load_frag(bs1);

        c00 = __builtin_amdgcn_wmma_f32_16x16x32_f16(false, a0, false, b0,
                                                     (short)0, c00, false, false);
        c01 = __builtin_amdgcn_wmma_f32_16x16x32_f16(false, a0, false, b1,
                                                     (short)0, c01, false, false);
        c10 = __builtin_amdgcn_wmma_f32_16x16x32_f16(false, a1, false, b0,
                                                     (short)0, c10, false, false);
        c11 = __builtin_amdgcn_wmma_f32_16x16x32_f16(false, a1, false, b1,
                                                     (short)0, c11, false, false);

        __syncthreads();                      // before next chunk overwrites Bs
    }

    // ---- epilogue: bias + activation, stage in LDS, coalesced store ----
    const float bv0 = b0ok ? bias[ncol0] : 0.0f;
    const float bv1 = b1ok ? bias[ncol1] : 0.0f;

    if (OUT_F32) {
        float* st = (float*)smem;
        auto put = [&](const v8f& c, int rb, int cl, float bv) {
#pragma unroll
            for (int r = 0; r < 8; ++r) {
                float v = c[r] + bv;
                if (GELU_ACT) v = gelu_f(v);
                st[(rb + 8 * g + r) * 64 + cl] = v;
            }
        };
        put(c00, wm * 32,      wn * 32 + l15,      bv0);
        put(c01, wm * 32,      wn * 32 + 16 + l15, bv1);
        put(c10, wm * 32 + 16, wn * 32 + l15,      bv0);
        put(c11, wm * 32 + 16, wn * 32 + 16 + l15, bv1);
        __syncthreads();
#pragma unroll
        for (int j = 0; j < 8; ++j) {         // 2048 chunks of 4 floats
            const int cchunk = t + 256 * j;
            const int row = cchunk >> 4;
            const int ch  = cchunk & 15;
            const int col0 = n_blk + ch * 4;  // Nvalid (300) is 4-aligned
            if (col0 < Nvalid) {
                *(float4*)((float*)out +
                           (size_t)(blockIdx.x * 128 + row) * Nout + col0) =
                    *(const float4*)&st[row * 64 + ch * 4];
            }
        }
    } else {
        _Float16* st = (_Float16*)smem;
        auto put = [&](const v8f& c, int rb, int cl, float bv) {
#pragma unroll
            for (int r = 0; r < 8; ++r) {
                float v = c[r] + bv;
                if (GELU_ACT) v = gelu_f(v);
                st[(rb + 8 * g + r) * 64 + cl] = (_Float16)v;
            }
        };
        put(c00, wm * 32,      wn * 32 + l15,      bv0);
        put(c01, wm * 32,      wn * 32 + 16 + l15, bv1);
        put(c10, wm * 32 + 16, wn * 32 + l15,      bv0);
        put(c11, wm * 32 + 16, wn * 32 + 16 + l15, bv1);
        __syncthreads();
#pragma unroll
        for (int j = 0; j < 4; ++j) {         // 1024 chunks of 8 halves
            const int cchunk = t + 256 * j;
            const int row = cchunk >> 3;
            const int ch  = cchunk & 7;
            *(v8h*)((_Float16*)out +
                    (size_t)(blockIdx.x * 128 + row) * Nout + n_blk + ch * 8) =
                *(const v8h*)&st[row * 64 + ch * 8];
        }
    }
}

// ---------------------------------------------------------------------------
// Launch
// ---------------------------------------------------------------------------
extern "C" void kernel_launch(void* const* d_in, const int* in_sizes, int n_in,
                              void* d_out, int out_size, void* d_ws, size_t ws_size,
                              hipStream_t stream) {
    const float* x   = (const float*)d_in[0];
    const float* bw1 = (const float*)d_in[1];
    const float* bb1 = (const float*)d_in[2];
    const float* bw2 = (const float*)d_in[3];
    const float* bb2 = (const float*)d_in[4];
    const float* wc1 = (const float*)d_in[5];
    const float* bc1 = (const float*)d_in[6];
    const float* wc2 = (const float*)d_in[7];
    const float* bc2 = (const float*)d_in[8];
    const float* wc3 = (const float*)d_in[9];
    const float* bc3 = (const float*)d_in[10];
    float* out = (float*)d_out;

    // workspace carve-out (all slices 16B-aligned)
    char* ws = (char*)d_ws;
    _Float16* f   = (_Float16*)ws;  ws += (size_t)B_SZ * KF  * sizeof(_Float16);
    _Float16* z1  = (_Float16*)ws;  ws += (size_t)B_SZ * C1D * sizeof(_Float16);
    _Float16* z2  = (_Float16*)ws;  ws += (size_t)B_SZ * C2D * sizeof(_Float16);
    _Float16* w1h = (_Float16*)ws;  ws += (size_t)C1D  * KF  * sizeof(_Float16);
    _Float16* w2h = (_Float16*)ws;  ws += (size_t)C2D  * C1D * sizeof(_Float16);
    _Float16* w3h = (_Float16*)ws;  ws += (size_t)OUTD * C2D * sizeof(_Float16);
    if ((size_t)(ws - (char*)d_ws) > ws_size) return;  // ws too small: bail

    // 1) convert combiner weights to fp16
    {
        int n1 = C1D * KF, n2 = C2D * C1D, n3 = OUTD * C2D;
        cvt_f16_kernel<<<(n1 + 255) / 256, 256, 0, stream>>>(wc1, w1h, n1);
        cvt_f16_kernel<<<(n2 + 255) / 256, 256, 0, stream>>>(wc2, w2h, n2);
        cvt_f16_kernel<<<(n3 + 255) / 256, 256, 0, stream>>>(wc3, w3h, n3);
    }

    // 2) all 1000 branch MLPs via padded WMMA; f = [B, 8000] fp16
    branch_kernel<<<dim3(B_SZ / 16, N_BR / 8), 256, 0, stream>>>(
        x, bw1, bb1, bw2, bb2, f);

    // 3) z1 = gelu(f @ wc1^T + bc1)      M=4096 N=256  K=8000
    gemm_wmma_kernel<true, false><<<dim3(B_SZ / 128, C1D / 64), 256, 0, stream>>>(
        f, w1h, bc1, z1, B_SZ, KF, C1D, C1D);

    // 4) z2 = gelu(z1 @ wc2^T + bc2)     M=4096 N=128  K=256
    gemm_wmma_kernel<true, false><<<dim3(B_SZ / 128, C2D / 64), 256, 0, stream>>>(
        z1, w2h, bc2, z2, B_SZ, C1D, C2D, C2D);

    // 5) out = z2 @ wc3^T + bc3          M=4096 N=300  K=128 (fp32, guarded)
    gemm_wmma_kernel<false, true><<<dim3(B_SZ / 128, (OUTD + 63) / 64), 256, 0, stream>>>(
        z2, w3h, bc3, out, B_SZ, C2D, OUTD, OUTD);
}
